// GSNN_11106785427524
// MI455X (gfx1250) — matrix operand order repaired and checked by reading, not compiled, not run on hardware
//
#include <hip/hip_runtime.h>
#include <hip/hip_bf16.h>
#include <math.h>

typedef __attribute__((ext_vector_type(16))) __bf16 v16bf;
typedef __attribute__((ext_vector_type(8)))  float  v8f;

#define N_NODES 100000
#define E_EDGES 1600000

static __device__ __forceinline__ __bf16 to_bf16(float f) { return (__bf16)f; }

// ---------------------------------------------------------------- fill zero
__global__ void k_fill_zero(float* __restrict__ p, long n) {
  long i = (long)blockIdx.x * blockDim.x + threadIdx.x;
  if (i < n) p[i] = 0.0f;
}

// ---------------------------------------------------------------- elementwise mul (dropout)
__global__ void k_mul_ew(const float* __restrict__ a, const float* __restrict__ b,
                         float* __restrict__ o, long n) {
  long i = (long)blockIdx.x * blockDim.x + threadIdx.x;
  if (i < n) o[i] = a[i] * b[i];
}

// ---------------------------------------------------------------- SpMM scatter: out[row] += w * h[col]
// one thread handles 4 consecutive columns of one edge; h row stride ldh, out row stride D
template <int D>
__global__ void k_spmm_scatter(const int* __restrict__ rows, const int* __restrict__ cols,
                               const float* __restrict__ w,
                               const float* __restrict__ h, int ldh,
                               float* __restrict__ out) {
  const int CH = D / 4;
  long idx = (long)blockIdx.x * blockDim.x + threadIdx.x;
  long total = (long)E_EDGES * CH;
  if (idx >= total) return;
  int e = (int)(idx / CH);
  int c = (int)(idx % CH) * 4;
  int r   = rows[e];
  int col = cols[e];
  float ww = w[e];
  const float4 v = *(const float4*)(h + (long)col * ldh + c);
  float* op = out + (long)r * D + c;
  atomicAdd(op + 0, ww * v.x);
  atomicAdd(op + 1, ww * v.y);
  atomicAdd(op + 2, ww * v.z);
  atomicAdd(op + 3, ww * v.w);
}

// ---------------------------------------------------------------- bf16 WMMA GEMM
// C = act(A[M x K] @ B[K x Dout] + bias), K = 32*KSTEPS, one wave per block.
// Each wave owns one 16-wide N tile (blockIdx.y) and sweeps MT=5 consecutive
// 16-row M tiles (blockIdx.x), register-caching the full B panel (KSTEPS v16bf
// fragments) and the bias across the sweep.
// A frag (ISA 16-bit A layout): lane<16 -> M=lane; VGPR i<4: K=8*half+2i, i>=4: K=16+8*half+2(i-4)
// B frag (ISA 16-bit B layout): N=lane&15; VGPR v<4: K=8*half+2v, v>=4: K=16+8*half+2(v-4)
template <bool RELU, int KSTEPS>
__global__ void k_wmma_gemm(const float* __restrict__ A, int lda,
                            const float* __restrict__ B, int ldb,
                            const float* __restrict__ bias,
                            float* __restrict__ C, int ldc, int mtiles_total) {
  const int lane  = threadIdx.x & 31;
  const int lidx  = lane & 15;
  const int lhalf = lane >> 4;
  const int n0    = blockIdx.y * 16;
  constexpr int MT = 5;

  // ---- register-resident B panel (K x 16), converted to bf16 once ----
  v16bf bfrag[KSTEPS];
  {
    const float* Bcol = B + n0 + lidx;
#pragma unroll
    for (int ks = 0; ks < KSTEPS; ++ks) {
#pragma unroll
      for (int v = 0; v < 8; ++v) {
        const int kk = ks * 32 + ((v < 4) ? (8 * lhalf + 2 * v) : (16 + 8 * lhalf + 2 * (v - 4)));
        bfrag[ks][2 * v]     = to_bf16(Bcol[(long)kk * ldb]);
        bfrag[ks][2 * v + 1] = to_bf16(Bcol[(long)(kk + 1) * ldb]);
      }
    }
  }
  const float bn = bias[n0 + lidx];

  for (int t = 0; t < MT; ++t) {
    const long mtile = (long)blockIdx.x * MT + t;
    if (mtile >= mtiles_total) break;
    const float* Arow = A + (mtile * 16 + lidx) * (long)lda;
    if (t + 1 < MT) __builtin_prefetch(Arow + 16 * (long)lda, 0, 0);  // global_prefetch next tile

    v8f acc = {};
#pragma unroll
    for (int ks = 0; ks < KSTEPS; ++ks) {
      v16bf a;
#pragma unroll
      for (int i = 0; i < 8; ++i) {
        const int kk = ks * 32 +
                       ((i < 4) ? (8 * lhalf + 2 * i) : (16 + 8 * lhalf + 2 * (i - 4)));
        const float2 av = *(const float2*)(Arow + kk);
        a[2 * i]     = to_bf16(av.x);
        a[2 * i + 1] = to_bf16(av.y);
      }
      acc = __builtin_amdgcn_wmma_f32_16x16x32_bf16(false, a, false, bfrag[ks], (short)0,
                                                    acc, false, false);
    }

#pragma unroll
    for (int r = 0; r < 8; ++r) {
      float v = acc[r] + bn;
      if (RELU) v = fmaxf(v, 0.0f);
      C[(mtile * 16 + r + 8 * lhalf) * (long)ldc + n0 + lidx] = v;
    }
  }
}

// ---------------------------------------------------------------- gumbel softmax (hard) + label select
// one wave per node, lane = class (Y_DIM == 32 == wave32)
__global__ void k_gumbel(const float* __restrict__ y_enc, const float* __restrict__ u,
                         const unsigned char* __restrict__ non_label,
                         const float* __restrict__ y_in,
                         float* __restrict__ out, int ldo) {
  const int wave = blockIdx.x * (blockDim.x >> 5) + (threadIdx.x >> 5);
  const int lane = threadIdx.x & 31;
  if (wave >= N_NODES) return;
  const long base = (long)wave * 32 + lane;
  const float g = -logf(-logf(u[base] + 1e-10f) + 1e-10f);
  const float v = y_enc[base] + g;  // tau = 1
  float m = v;
#pragma unroll
  for (int o = 16; o > 0; o >>= 1) m = fmaxf(m, __shfl_xor(m, o, 32));
  const float e = expf(v - m);
  float s = e;
#pragma unroll
  for (int o = 16; o > 0; o >>= 1) s += __shfl_xor(s, o, 32);
  const float soft = e / s;
  int idx = (v == m) ? lane : 32;
#pragma unroll
  for (int o = 16; o > 0; o >>= 1) { int t = __shfl_xor(idx, o, 32); idx = idx < t ? idx : t; }
  const float hard = (lane == idx) ? 1.0f : 0.0f;
  const float yg = (hard + soft) - soft;  // straight-through forward value
  out[(long)wave * ldo + lane] = non_label[wave] ? yg : y_in[base];
}

// ---------------------------------------------------------------- column-sum accumulate (128 cols)
__global__ void k_colsum(const float* __restrict__ A, int ld, int rows_per_block,
                         float* __restrict__ colsum) {
  const int c = threadIdx.x;  // blockDim.x == 128
  long r0 = (long)blockIdx.x * rows_per_block;
  long r1 = r0 + rows_per_block; if (r1 > N_NODES) r1 = N_NODES;
  float s = 0.0f;
  for (long r = r0; r < r1; ++r) s += A[r * ld + c];
  atomicAdd(&colsum[c], s);
}

// ---------------------------------------------------------------- latent path (1 block, 128 threads)
// mean -> r_graph = m@W_hr+b -> hr = relu(rg@W_rh+b) -> mu/sigma -> z
__global__ void k_latent(const float* __restrict__ colsum,
                         const float* __restrict__ W_hr, const float* __restrict__ b_hr,
                         const float* __restrict__ W_rh, const float* __restrict__ b_rh,
                         const float* __restrict__ W_mu, const float* __restrict__ b_mu,
                         const float* __restrict__ W_sig, const float* __restrict__ b_sig,
                         const float* __restrict__ z_eps, float* __restrict__ z_out) {
  __shared__ float sm[128], srg[128], shr[128];
  const int t = threadIdx.x;
  sm[t] = colsum[t] * (1.0f / N_NODES);
  __syncthreads();
  float acc = b_hr[t];
  for (int i = 0; i < 128; ++i) acc += sm[i] * W_hr[i * 128 + t];
  srg[t] = acc;
  __syncthreads();
  acc = b_rh[t];
  for (int i = 0; i < 128; ++i) acc += srg[i] * W_rh[i * 128 + t];
  shr[t] = fmaxf(acc, 0.0f);
  __syncthreads();
  if (t < 64) {
    float mu = b_mu[t], sv = b_sig[t];
    for (int i = 0; i < 128; ++i) {
      mu += shr[i] * W_mu[i * 64 + t];
      sv += shr[i] * W_sig[i * 64 + t];
    }
    const float sigma = 0.1f + 0.9f / (1.0f + expf(-sv));
    z_out[t] = mu + sigma * z_eps[t];
  }
}

// ---------------------------------------------------------------- append z + L2 row-normalize [N,192]
// one wave per row: cols 0..127 already hold relu(xd@W_xh+b); cols 128..191 <- z; row /= (||row||+1e-6)
__global__ void k_append_z_norm(float* __restrict__ h, const float* __restrict__ z) {
  const int wave = blockIdx.x * (blockDim.x >> 5) + (threadIdx.x >> 5);
  const int lane = threadIdx.x & 31;
  if (wave >= N_NODES) return;
  float* row = h + (long)wave * 192;
  float vals[6];
  float ss = 0.0f;
#pragma unroll
  for (int i = 0; i < 6; ++i) {
    const int c = i * 32 + lane;
    const float v = (c < 128) ? row[c] : z[c - 128];
    vals[i] = v;
    ss += v * v;
  }
#pragma unroll
  for (int o = 16; o > 0; o >>= 1) ss += __shfl_xor(ss, o, 32);
  const float inv = 1.0f / (sqrtf(ss) + 1e-6f);
#pragma unroll
  for (int i = 0; i < 6; ++i) row[i * 32 + lane] = vals[i] * inv;
}

// ================================================================ launch
extern "C" void kernel_launch(void* const* d_in, const int* in_sizes, int n_in,
                              void* d_out, int out_size, void* d_ws, size_t ws_size,
                              hipStream_t stream) {
  (void)in_sizes; (void)n_in; (void)out_size; (void)ws_size;
  const float* x        = (const float*)d_in[0];
  const float* y_in     = (const float*)d_in[1];
  const int*   ei       = (const int*)d_in[2];
  const float* ew       = (const float*)d_in[3];
  const unsigned char* non_label = (const unsigned char*)d_in[4];
  const float* dropout  = (const float*)d_in[5];
  const float* u_gum    = (const float*)d_in[6];
  const float* z_eps    = (const float*)d_in[7];
  const float* Wg1 = (const float*)d_in[8],  *bg1 = (const float*)d_in[9];
  const float* Wg2 = (const float*)d_in[10], *bg2 = (const float*)d_in[11];
  const float* W_xy = (const float*)d_in[12], *b_xy = (const float*)d_in[13];
  const float* W_hr = (const float*)d_in[14], *b_hr = (const float*)d_in[15];
  const float* W_rh = (const float*)d_in[16], *b_rh = (const float*)d_in[17];
  const float* W_mu = (const float*)d_in[18], *b_mu = (const float*)d_in[19];
  const float* W_sig = (const float*)d_in[20], *b_sig = (const float*)d_in[21];
  const float* W_xh = (const float*)d_in[22], *b_xh = (const float*)d_in[23];
  const float* W_h2 = (const float*)d_in[24], *b_h2 = (const float*)d_in[25];
  const float* W_hy = (const float*)d_in[26], *b_hy = (const float*)d_in[27];
  float* out = (float*)d_out;

  const int* e_rows = ei;              // edge_index[0]
  const int* e_cols = ei + E_EDGES;    // edge_index[1]

  // -------- workspace layout (floats) --------
  float* wsf    = (float*)d_ws;
  float* colsum = wsf;                 // 128
  float* zbuf   = wsf + 128;           // 64
  float* S1     = wsf + 1024;                       // [N,128]
  float* B160   = S1 + (long)N_NODES * 128;         // [N,160]
  float* B192   = B160 + (long)N_NODES * 160;       // [N,192]
  float* S2     = B192 + (long)N_NODES * 192;       // [N,192]

  const long n128 = (long)N_NODES * 128;
  const long n192 = (long)N_NODES * 192;
  const dim3 blk256(256);
  const int  MTILES = N_NODES / 16;            // 6250 exactly
  const int  GX     = (MTILES + 4) / 5;        // MT=5 -> 1250 exactly
  const long sp128 = (long)E_EDGES * (128 / 4);
  const long sp192 = (long)E_EDGES * (192 / 4);
  const int  nwaveblk = (N_NODES + 7) / 8;     // 8 waves / 256-thread block

  // 1) S1 = spmm(x)
  k_fill_zero<<<(n128 + 255) / 256, blk256, 0, stream>>>(S1, n128);
  k_spmm_scatter<128><<<(sp128 + 255) / 256, blk256, 0, stream>>>(e_rows, e_cols, ew, x, 128, S1);
  // 2) B160[:, :128] = relu(S1 @ Wg1 + bg1), K=128
  k_wmma_gemm<true, 4><<<dim3(GX, 8), dim3(32), 0, stream>>>(S1, 128, Wg1, 128, bg1, B160, 160, MTILES);
  // 3) S1 = spmm(h_emb)
  k_fill_zero<<<(n128 + 255) / 256, blk256, 0, stream>>>(S1, n128);
  k_spmm_scatter<128><<<(sp128 + 255) / 256, blk256, 0, stream>>>(e_rows, e_cols, ew, B160, 160, S1);
  // 4) y_encode (into S2 scratch) = S1 @ Wg2 + bg2, K=128
  k_wmma_gemm<false, 4><<<dim3(GX, 2), dim3(32), 0, stream>>>(S1, 128, Wg2, 32, bg2, S2, 32, MTILES);
  // 5) gumbel-hard + label select -> B160[:, 128:160]
  k_gumbel<<<nwaveblk, blk256, 0, stream>>>(S2, u_gum, non_label, y_in, B160 + 128, 160);
  // 6) h_enc (S1) = relu(B160 @ W_xy + b_xy), K=160
  k_wmma_gemm<true, 5><<<dim3(GX, 8), dim3(32), 0, stream>>>(B160, 160, W_xy, 128, b_xy, S1, 128, MTILES);
  // 7) readout: colmean(h_enc) -> latent z
  k_fill_zero<<<1, dim3(128), 0, stream>>>(colsum, 128);
  k_colsum<<<(N_NODES + 255) / 256, dim3(128), 0, stream>>>(S1, 128, 256, colsum);
  k_latent<<<1, dim3(128), 0, stream>>>(colsum, W_hr, b_hr, W_rh, b_rh, W_mu, b_mu,
                                        W_sig, b_sig, z_eps, zbuf);
  // 8) xd = x * dropout (reuse S1)
  k_mul_ew<<<(n128 + 255) / 256, blk256, 0, stream>>>(x, dropout, S1, n128);
  // 9) B192[:, :128] = relu(xd @ W_xh + b_xh), K=128
  k_wmma_gemm<true, 4><<<dim3(GX, 8), dim3(32), 0, stream>>>(S1, 128, W_xh, 128, b_xh, B192, 192, MTILES);
  // 10) append z, L2-normalize rows of B192
  k_append_z_norm<<<nwaveblk, blk256, 0, stream>>>(B192, zbuf);
  // 11) S2 = spmm(B192), D=192
  k_fill_zero<<<(n192 + 255) / 256, blk256, 0, stream>>>(S2, n192);
  k_spmm_scatter<192><<<(sp192 + 255) / 256, blk256, 0, stream>>>(e_rows, e_cols, ew, B192, 192, S2);
  // 12) B192 = relu(S2 @ W_h2 + b_h2), K=192
  k_wmma_gemm<true, 6><<<dim3(GX, 12), dim3(32), 0, stream>>>(S2, 192, W_h2, 192, b_h2, B192, 192, MTILES);
  // 13) S2 = spmm(B192)
  k_fill_zero<<<(n192 + 255) / 256, blk256, 0, stream>>>(S2, n192);
  k_spmm_scatter<192><<<(sp192 + 255) / 256, blk256, 0, stream>>>(e_rows, e_cols, ew, B192, 192, S2);
  // 14) y_pred = S2 @ W_hy + b_hy, K=192
  k_wmma_gemm<false, 6><<<dim3(GX, 2), dim3(32), 0, stream>>>(S2, 192, W_hy, 32, b_hy, out, 32, MTILES);
}
